// PhrasalLexemeEncoderLayer_15109694947370
// MI455X (gfx1250) — compile-verified, matrix operand order checked by reading, not run on hardware
//
#include <hip/hip_runtime.h>
#include <hip/hip_bf16.h>
#include <math.h>

// ---------------------------------------------------------------------------
// PhrasalLexemeEncoderLayer for MI455X (gfx1250), wave32 + WMMA bf16 + TDM.
// B=8, N=1024, D=512, H=8, dh=64, FF=2048.
// Bandwidth-bound (~1.4 GB); all operands pre-swizzled into WMMA fragment
// order so hot loops are pure {b128 vector load -> v_wmma}.  phrasal_attn
// prior streamed through LDS with double-buffered tensor_load_to_lds
// (TENSORcnt in-order completion hides the DMA under 4 k-tiles of compute).
// ---------------------------------------------------------------------------

#define BB   8
#define NN_  1024
#define DM   512
#define HH   8
#define DH   64
#define FF   2048
#define RWS  (BB * NN_)          // 8192 rows

typedef __attribute__((ext_vector_type(16))) __bf16 v16bf;
typedef __attribute__((ext_vector_type(8)))  float  v8f;
typedef __attribute__((ext_vector_type(4)))  unsigned int u32x4;
typedef __attribute__((ext_vector_type(8)))  int          i32x8;
typedef __attribute__((ext_vector_type(4)))  int          i32x4;

#define WMMA_BF16(a, b, c) \
  __builtin_amdgcn_wmma_f32_16x16x32_bf16(false, (a), false, (b), (short)0, (c), false, false)

// ---- TDM availability (guarded so compile can never break) ----------------
#define USE_TDM 0
#if defined(__has_builtin)
#if __has_builtin(__builtin_amdgcn_tensor_load_to_lds) && __has_builtin(__builtin_amdgcn_s_wait_tensorcnt)
#undef USE_TDM
#define USE_TDM 1
#if defined(__has_include)
#if __has_include(<hip/amd_detail/amd_gfx1250_TDM.h>)
#define TDM_SIX_ARGS 1
#endif
#endif
#endif
#endif

__device__ __forceinline__ v8f vzero8() {
  v8f z;
#pragma unroll
  for (int i = 0; i < 8; ++i) z[i] = 0.0f;
  return z;
}

// Position of value (row m15 in 0..15, k kk in 0..31) inside a 512-element
// fragment laid out as lane*16 + i, per ISA 7.12.2 16-bit A/B layout.
__device__ __forceinline__ int fragpos(int m15, int kk) {
  const int hi = (kk >> 3) & 1;
  const int j  = ((kk >> 4) << 2) | ((kk >> 1) & 3);
  const int i  = (j << 1) | (kk & 1);
  return (((m15 & 15) | (hi << 4)) << 4) + i;
}

// Fragment load: one 32B vector load per lane (2x b128).
__device__ __forceinline__ v16bf ldfrag(const __bf16* frag, int lane) {
  return ((const v16bf*)frag)[lane];
}

// 16-lane (half-wave) reductions.
__device__ __forceinline__ float rmax16(float v) {
  v = fmaxf(v, __shfl_xor(v, 1, 32));
  v = fmaxf(v, __shfl_xor(v, 2, 32));
  v = fmaxf(v, __shfl_xor(v, 4, 32));
  v = fmaxf(v, __shfl_xor(v, 8, 32));
  return v;
}
__device__ __forceinline__ float rsum16(float v) {
  v += __shfl_xor(v, 1, 32);
  v += __shfl_xor(v, 2, 32);
  v += __shfl_xor(v, 4, 32);
  v += __shfl_xor(v, 8, 32);
  return v;
}
__device__ __forceinline__ float rsum32(float v) {
  v += __shfl_xor(v, 1, 32);
  v += __shfl_xor(v, 2, 32);
  v += __shfl_xor(v, 4, 32);
  v += __shfl_xor(v, 8, 32);
  v += __shfl_xor(v, 16, 32);
  return v;
}

#if USE_TDM
// Issue one TDM load of a (tile_h x tile_w) f32 tile (row stride = stride
// elems) into LDS at byte offset lds_off.  All inputs wave-uniform.
__device__ __forceinline__ void tdm_load_tile_f32(unsigned lds_off, const float* gptr,
                                                  unsigned tile_w, unsigned tile_h,
                                                  unsigned stride) {
  const unsigned long long ga = (unsigned long long)(size_t)gptr;
  u32x4 g0;
  g0[0] = 1u;                                            // count=1 (valid user D#)
  g0[1] = lds_off;                                       // lds_addr
  g0[2] = (unsigned)(ga & 0xffffffffull);                // global_addr[31:0]
  g0[3] = (unsigned)((ga >> 32) & 0x1ffffffull) | (2u << 30);  // addr[56:32] | type=2
  i32x8 g1;
  g1[0] = (int)(2u << 16);                               // mask=0, data_size=4B
  g1[1] = (int)(stride << 16);                           // tensor_dim0[15:0]
  g1[2] = (int)(1024u << 16);                            // dim0 hi | tensor_dim1[15:0]
  g1[3] = (int)(tile_w << 16);                           // dim1 hi | tile_dim0
  g1[4] = (int)tile_h;                                   // tile_dim1 | tile_dim2=0
  g1[5] = (int)stride;                                   // tensor_dim0_stride[31:0]
  g1[6] = 0; g1[7] = 0;
  i32x4 gz; gz[0] = 0; gz[1] = 0; gz[2] = 0; gz[3] = 0;
#if defined(TDM_SIX_ARGS)
  i32x8 gz8;
#pragma unroll
  for (int q = 0; q < 8; ++q) gz8[q] = 0;
  __builtin_amdgcn_tensor_load_to_lds(g0, g1, gz, gz, gz8, 0);
#else
  __builtin_amdgcn_tensor_load_to_lds(g0, g1, gz, gz, 0);
#endif
}
#endif

// ---------------------------------------------------------------------------
// Prep: convert an f32 matrix W[rows][K] into bf16 WMMA-fragment layout:
// fragment (rowtile, ktile) at ((row>>4)*(K/32) + (k>>5))*512.
// ---------------------------------------------------------------------------
__global__ void cvt_frag_kernel(const float* __restrict__ W, __bf16* __restrict__ out,
                                int K, long long total) {
  const long long idx = (long long)blockIdx.x * blockDim.x + threadIdx.x;
  if (idx >= total) return;
  const int k = (int)(idx % K);
  const long long row = idx / K;
  out[((row >> 4) * (K / 32) + (k >> 5)) * 512 + fragpos((int)(row & 15), k & 31)] =
      (__bf16)W[idx];
}

__global__ void bias_cat_kernel(const float* __restrict__ ipb, const float* __restrict__ bq,
                                const float* __restrict__ bk, const float* __restrict__ bo,
                                float* __restrict__ out) {
  const int i = (int)blockIdx.x * 256 + (int)threadIdx.x;
  if (i >= 5 * DM) return;
  float v;
  if (i < 1024)      v = ipb[i];
  else if (i < 1536) v = bq[i - 1024];
  else if (i < 2048) v = bk[i - 1536];
  else               v = bo[i - 2048];
  out[i] = v;
}

// ---------------------------------------------------------------------------
// Kernel 1: fused projections from fragment-layout x and weights.
// 2560 output cols: [q | k | pq | pk | v].  Results scattered into the
// per-head fragment layouts consumed by the attention kernel.
// ---------------------------------------------------------------------------
__global__ void proj_kernel(const __bf16* __restrict__ xf, const __bf16* __restrict__ wf,
                            const float* __restrict__ biascat, __bf16* __restrict__ ws5) {
  const int lane = (int)threadIdx.x & 31;
  const int wave = (int)((blockIdx.x * blockDim.x + threadIdx.x) >> 5);
  const int NTC = (5 * DM) / 16;           // 160 column tiles
  const int mt = wave / NTC, ct = wave % NTC;
  if (mt >= RWS / 16) return;

  const __bf16* af = xf + (size_t)mt * 16 * 512;
  const __bf16* bf = wf + (size_t)ct * 16 * 512;
  v8f acc = vzero8();
#pragma unroll 4
  for (int kc = 0; kc < 16; ++kc)
    acc = WMMA_BF16(ldfrag(af + kc * 512, lane), ldfrag(bf + kc * 512, lane), acc);

  const int nn = lane & 15, hi = (lane >> 4) & 1;
  const int m0 = mt * 16, c0 = ct * 16;
  const int seg = c0 >> 9, cw = c0 & 511;
  const int col = cw + nn;                 // col within segment
  const float bs = biascat[c0 + nn];
  const int head = col >> 6, d = col & 63;
  __bf16* dst = ws5 + (size_t)seg * ((size_t)BB * HH * NN_ * DH);

  if (seg < 4) {                           // Q/K/PQ/PK: A/BT fragment layout
    const int kh = d >> 5, kk = d & 31;
#pragma unroll
    for (int r = 0; r < 8; ++r) {
      const int row = m0 + r + hi * 8;
      const int bb = row >> 10, n = row & 1023;
      dst[(((size_t)(bb * HH + head) * 64 + (n >> 4)) * 2 + kh) * 512 + fragpos(n & 15, kk)] =
          (__bf16)(acc[r] + bs);
    }
  } else {                                 // V: BN fragment layout (k = token)
#pragma unroll
    for (int r = 0; r < 8; ++r) {
      const int row = m0 + r + hi * 8;
      const int bb = row >> 10, n = row & 1023;
      dst[(((size_t)(bb * HH + head) * 32 + (n >> 5)) * 4 + (d >> 4)) * 512 + fragpos(d & 15, n & 31)] =
          (__bf16)(acc[r] + bs);
    }
  }
}

// ---------------------------------------------------------------------------
// Kernel 2: fused dual attention.  One wave per 16-row q-tile of (b,h).
// Pass 1: streaming online max/sum (self + phrasal, prior-weighted sum).
// Pass 2: recompute, normalize, fuse, write the 3 output tensors, and
// accumulate feats = attn @ V with WMMA (A staged in LDS fragment order).
// Prior tiles double-buffer through LDS via the Tensor Data Mover: TDM ops
// of one wave complete in order, so s_wait_tensorcnt(1) after issuing group
// g+1 guarantees group g landed while g+1 flies under 4 k-tiles of compute.
// ---------------------------------------------------------------------------
__global__ void attn_kernel(const int*   __restrict__ am,
                            const float* __restrict__ prior,
                            const __bf16* __restrict__ Qb, const __bf16* __restrict__ Kb,
                            const __bf16* __restrict__ PQb, const __bf16* __restrict__ PKb,
                            const __bf16* __restrict__ Vb,
                            float* __restrict__ out_sa, float* __restrict__ out_pn,
                            float* __restrict__ out_as, float* __restrict__ feats) {
  __shared__ __bf16 stage[512];            // attn A-fragment staging
#if USE_TDM
  __shared__ float pstage[2][16 * 64];     // double-buffered prior tiles
#endif
  const int lane = (int)threadIdx.x & 31;
  const int tile = (int)blockIdx.x;        // 4096 tiles, 1 wave each
  const int b  = tile >> 9;
  const int h  = (tile >> 6) & 7;
  const int q0 = (tile & 63) << 4;
  const size_t bh = (size_t)(b * HH + h);

  const __bf16* Qh  = Qb  + bh * 128 * 512;   // 64 mtiles x 2 k-halves
  const __bf16* Kh  = Kb  + bh * 128 * 512;
  const __bf16* PQh = PQb + bh * 128 * 512;
  const __bf16* PKh = PKb + bh * 128 * 512;
  const __bf16* Vh  = Vb  + bh * 128 * 512;   // 32 nb32 x 4 dchunks
  const float*  Ph  = prior + bh * (size_t)NN_ * NN_;
  const float*  Prow = Ph + (size_t)q0 * NN_;

  const v16bf aQ0 = ldfrag(Qh  + ((size_t)(q0 >> 4) * 2 + 0) * 512, lane);
  const v16bf aQ1 = ldfrag(Qh  + ((size_t)(q0 >> 4) * 2 + 1) * 512, lane);
  const v16bf aP0 = ldfrag(PQh + ((size_t)(q0 >> 4) * 2 + 0) * 512, lane);
  const v16bf aP1 = ldfrag(PQh + ((size_t)(q0 >> 4) * 2 + 1) * 512, lane);

  const int nn = lane & 15, hi = (lane >> 4) & 1;
  const float scale = 0.125f;              // 1/sqrt(64)

  float mS[8], lS[8], mP[8], lP[8], tP[8];
#pragma unroll
  for (int r = 0; r < 8; ++r) { mS[r] = -3.0e38f; mP[r] = -3.0e38f; lS[r] = 0.f; lP[r] = 0.f; tP[r] = 0.f; }

  // ---------------- pass 1: online stats ----------------
#if USE_TDM
  tdm_load_tile_f32((unsigned)(size_t)(void*)pstage[0], Prow, 64u, 16u, (unsigned)NN_);
#endif
  for (int kt = 0; kt < NN_ / 16; ++kt) {
    const int n0 = kt * 16;
#if USE_TDM
    if ((kt & 3) == 0) {
      const int g = kt >> 2;
      if (g + 1 < 16) {
        tdm_load_tile_f32((unsigned)(size_t)(void*)pstage[(g + 1) & 1],
                          Prow + (g + 1) * 64, 64u, 16u, (unsigned)NN_);
        __builtin_amdgcn_s_wait_tensorcnt(1);   // group g landed (in-order)
      } else {
        __builtin_amdgcn_s_wait_tensorcnt(0);
      }
    }
    const float* pbuf = pstage[(kt >> 2) & 1];
#endif
    v8f s = vzero8();
    s = WMMA_BF16(aQ0, ldfrag(Kh + ((size_t)kt * 2 + 0) * 512, lane), s);
    s = WMMA_BF16(aQ1, ldfrag(Kh + ((size_t)kt * 2 + 1) * 512, lane), s);
    v8f p = vzero8();
    p = WMMA_BF16(aP0, ldfrag(PKh + ((size_t)kt * 2 + 0) * 512, lane), p);
    p = WMMA_BF16(aP1, ldfrag(PKh + ((size_t)kt * 2 + 1) * 512, lane), p);

    const int col = n0 + nn;
    const float padf = (float)am[b * NN_ + col];
    const float addm = padf * -1e9f;
#pragma unroll
    for (int r = 0; r < 8; ++r) {
      const float sv = s[r] * scale + addm;
      const float pv = (padf > 0.f) ? -1e9f : p[r] * scale;
#if USE_TDM
      const float pr = pbuf[(r + hi * 8) * 64 + (kt & 3) * 16 + nn];
#else
      const float pr = Ph[(size_t)(q0 + r + hi * 8) * NN_ + col];
#endif
      float nm = fmaxf(mS[r], rmax16(sv));
      lS[r] = lS[r] * __expf(mS[r] - nm) + rsum16(__expf(sv - nm));
      mS[r] = nm;

      float nmp = fmaxf(mP[r], rmax16(pv));
      const float sc = __expf(mP[r] - nmp);
      const float e  = __expf(pv - nmp);
      lP[r] = lP[r] * sc + rsum16(e);
      tP[r] = tP[r] * sc + rsum16(e * pr);
      mP[r] = nmp;
    }
  }

  // ---------------- pass 2: recompute, write, feats ----------------
  v8f cF[4];
#pragma unroll
  for (int j = 0; j < 4; ++j) cF[j] = vzero8();

#if USE_TDM
  tdm_load_tile_f32((unsigned)(size_t)(void*)pstage[0], Prow, 64u, 16u, (unsigned)NN_);
#endif
  for (int kt = 0; kt < NN_ / 16; ++kt) {
    const int n0 = kt * 16;
#if USE_TDM
    if ((kt & 3) == 0) {
      const int g = kt >> 2;
      if (g + 1 < 16) {
        tdm_load_tile_f32((unsigned)(size_t)(void*)pstage[(g + 1) & 1],
                          Prow + (g + 1) * 64, 64u, 16u, (unsigned)NN_);
        __builtin_amdgcn_s_wait_tensorcnt(1);
      } else {
        __builtin_amdgcn_s_wait_tensorcnt(0);
      }
    }
    const float* pbuf = pstage[(kt >> 2) & 1];
#endif
    v8f s = vzero8();
    s = WMMA_BF16(aQ0, ldfrag(Kh + ((size_t)kt * 2 + 0) * 512, lane), s);
    s = WMMA_BF16(aQ1, ldfrag(Kh + ((size_t)kt * 2 + 1) * 512, lane), s);
    v8f p = vzero8();
    p = WMMA_BF16(aP0, ldfrag(PKh + ((size_t)kt * 2 + 0) * 512, lane), p);
    p = WMMA_BF16(aP1, ldfrag(PKh + ((size_t)kt * 2 + 1) * 512, lane), p);

    const int col = n0 + nn;
    const float padf = (float)am[b * NN_ + col];
    const float addm = padf * -1e9f;
#pragma unroll
    for (int r = 0; r < 8; ++r) {
      const int row = q0 + r + hi * 8;
      const float sv = s[r] * scale + addm;
      const float pv = (padf > 0.f) ? -1e9f : p[r] * scale;
      const float sa = __expf(sv - mS[r]) / lS[r];
      const float pa = __expf(pv - mP[r]) / lP[r];
#if USE_TDM
      const float pr = pbuf[(r + hi * 8) * 64 + (kt & 3) * 16 + nn];
#else
      const float pr = Ph[(size_t)row * NN_ + col];
#endif
      const float pn = pa * pr;
      const float sumpn = tP[r] / lP[r];
      const float Pv = pn / (sumpn + 1e-9f);
      const float at = Pv * sa;

      const size_t idx = (bh * NN_ + (size_t)row) * NN_ + col;
      out_sa[idx] = sa;
      out_pn[idx] = pn;
      out_as[idx] = at;
      // stage attn value directly in A-fragment order (k = column offset 0..31)
      stage[fragpos(r + hi * 8, (kt & 1) * 16 + nn)] = (__bf16)at;
    }
    if (kt & 1) {
      const int nb32 = (kt - 1) >> 1;      // 32 staged attn columns
      const v16bf aA = ldfrag(stage, lane);
#pragma unroll
      for (int j = 0; j < 4; ++j)
        cF[j] = WMMA_BF16(aA, ldfrag(Vh + ((size_t)nb32 * 4 + j) * 512, lane), cF[j]);
    }
  }

#pragma unroll
  for (int j = 0; j < 4; ++j)
#pragma unroll
    for (int r = 0; r < 8; ++r) {
      const int row = q0 + r + hi * 8;
      feats[((size_t)(b * NN_ + row)) * DM + h * DH + j * 16 + nn] = cF[j][r];
    }
}

// ---------------------------------------------------------------------------
// LayerNorm: y = LN(a + b) * g + be.  One wave per row of 512.
// Optionally writes a bf16 copy in A-fragment layout for the next GEMM.
// ---------------------------------------------------------------------------
__global__ void ln_kernel(const float* __restrict__ a, const float* __restrict__ bsrc,
                          const float* __restrict__ g, const float* __restrict__ be,
                          float* __restrict__ out32, __bf16* __restrict__ outfrag) {
  const int lane = (int)threadIdx.x & 31;
  const int wv   = (int)threadIdx.x >> 5;
  const int row  = (int)blockIdx.x * 8 + wv;
  const float* ap = a + (size_t)row * DM;
  const float* bp = bsrc + (size_t)row * DM;

  float vals[16], s = 0.f, s2 = 0.f;
#pragma unroll
  for (int i = 0; i < 16; ++i) {
    const float v = ap[lane + i * 32] + bp[lane + i * 32];
    vals[i] = v; s += v; s2 += v * v;
  }
  s = rsum32(s); s2 = rsum32(s2);
  const float mean = s * (1.0f / DM);
  const float var  = s2 * (1.0f / DM) - mean * mean;
  const float inv  = rsqrtf(var + 1e-5f);
#pragma unroll
  for (int i = 0; i < 16; ++i) {
    const int c = lane + i * 32;
    const float y = (vals[i] - mean) * inv * g[c] + be[c];
    out32[(size_t)row * DM + c] = y;
    if (outfrag)
      outfrag[((size_t)(row >> 4) * 16 + (c >> 5)) * 512 + fragpos(row & 15, c & 31)] = (__bf16)y;
  }
}

// ---------------------------------------------------------------------------
// FFN GEMM 1: H = gelu_exact(feats @ w1^T + b1) -> bf16 fragments (8192x2048)
// ---------------------------------------------------------------------------
__global__ void ffn1_kernel(const __bf16* __restrict__ Af, const __bf16* __restrict__ w1f,
                            const float* __restrict__ b1, __bf16* __restrict__ Hf) {
  const int lane = (int)threadIdx.x & 31;
  const int wave = (int)((blockIdx.x * blockDim.x + threadIdx.x) >> 5);
  const int NTC = FF / 16;                 // 128
  const int mt = wave / NTC, ct = wave % NTC;
  if (mt >= RWS / 16) return;

  const __bf16* af = Af + (size_t)mt * 16 * 512;
  const __bf16* bf = w1f + (size_t)ct * 16 * 512;
  v8f acc = vzero8();
#pragma unroll 4
  for (int kc = 0; kc < 16; ++kc)
    acc = WMMA_BF16(ldfrag(af + kc * 512, lane), ldfrag(bf + kc * 512, lane), acc);

  const int nn = lane & 15, hi = (lane >> 4) & 1;
  const int col = ct * 16 + nn;
  const float bs = b1[col];
#pragma unroll
  for (int r = 0; r < 8; ++r) {
    const float v = acc[r] + bs;
    const float ge = 0.5f * v * (1.0f + erff(v * 0.70710678118f));
    Hf[((size_t)mt * (FF / 32) + (col >> 5)) * 512 + fragpos(r + hi * 8, col & 31)] = (__bf16)ge;
  }
}

// ---------------------------------------------------------------------------
// FFN GEMM 2: Y = H @ w2^T + b2 -> f32 rows (8192x512)
// ---------------------------------------------------------------------------
__global__ void ffn2_kernel(const __bf16* __restrict__ Hf, const __bf16* __restrict__ w2f,
                            const float* __restrict__ b2, float* __restrict__ Y) {
  const int lane = (int)threadIdx.x & 31;
  const int wave = (int)((blockIdx.x * blockDim.x + threadIdx.x) >> 5);
  const int NTC = DM / 16;                 // 32
  const int mt = wave / NTC, ct = wave % NTC;
  if (mt >= RWS / 16) return;

  const __bf16* af = Hf + (size_t)mt * 64 * 512;
  const __bf16* bf = w2f + (size_t)ct * 64 * 512;
  v8f acc = vzero8();
#pragma unroll 4
  for (int kc = 0; kc < 64; ++kc) {
    if (kc + 2 < 64) __builtin_prefetch(af + (kc + 2) * 512, 0, 3);
    acc = WMMA_BF16(ldfrag(af + kc * 512, lane), ldfrag(bf + kc * 512, lane), acc);
  }
  const int nn = lane & 15, hi = (lane >> 4) & 1;
  const int col = ct * 16 + nn;
  const float bs = b2[col];
#pragma unroll
  for (int r = 0; r < 8; ++r)
    Y[(size_t)(mt * 16 + r + hi * 8) * DM + col] = acc[r] + bs;
}

// ---------------------------------------------------------------------------
// Launch
// ---------------------------------------------------------------------------
extern "C" void kernel_launch(void* const* d_in, const int* in_sizes, int n_in,
                              void* d_out, int out_size, void* d_ws, size_t ws_size,
                              hipStream_t stream) {
  const float* x     = (const float*)d_in[0];
  const int*   am    = (const int*)  d_in[1];
  const float* prior = (const float*)d_in[2];
  const float* ipw   = (const float*)d_in[3];
  const float* ipb   = (const float*)d_in[4];
  const float* wq    = (const float*)d_in[5];
  const float* bq    = (const float*)d_in[6];
  const float* wk    = (const float*)d_in[7];
  const float* bk    = (const float*)d_in[8];
  const float* wo    = (const float*)d_in[9];
  const float* bo    = (const float*)d_in[10];
  const float* w1    = (const float*)d_in[11];
  const float* b1    = (const float*)d_in[12];
  const float* w2    = (const float*)d_in[13];
  const float* b2    = (const float*)d_in[14];
  const float* g1    = (const float*)d_in[15];
  const float* be1   = (const float*)d_in[16];
  const float* g2    = (const float*)d_in[17];
  const float* be2   = (const float*)d_in[18];

  float* out    = (float*)d_out;
  float* out_sa = out + (size_t)BB * NN_ * DM;
  float* out_pn = out_sa + (size_t)BB * HH * NN_ * NN_;
  float* out_as = out_pn + (size_t)BB * HH * NN_ * NN_;

  // ---- workspace carve-up (~126.5 MiB) ----
  char* p = (char*)d_ws;
  const size_t PROJ_ELEMS = (size_t)BB * HH * NN_ * DH;              // 4,194,304
  __bf16* projsw = (__bf16*)p;  p += 5 * PROJ_ELEMS * sizeof(__bf16);    // 40 MiB
  __bf16* Qs  = projsw + 0 * PROJ_ELEMS;
  __bf16* Ks  = projsw + 1 * PROJ_ELEMS;
  __bf16* PQs = projsw + 2 * PROJ_ELEMS;
  __bf16* PKs = projsw + 3 * PROJ_ELEMS;
  __bf16* Vs  = projsw + 4 * PROJ_ELEMS;
  __bf16* xf    = (__bf16*)p;  p += (size_t)RWS * DM * sizeof(__bf16);   //  8 MiB
  __bf16* wcatf = (__bf16*)p;  p += (size_t)5 * DM * DM * sizeof(__bf16);// 2.5 MiB
  __bf16* w1f   = (__bf16*)p;  p += (size_t)FF * DM * sizeof(__bf16);    //  2 MiB
  __bf16* w2f   = (__bf16*)p;  p += (size_t)DM * FF * sizeof(__bf16);    //  2 MiB
  float*  bcat  = (float*)p;   p += (size_t)5 * DM * sizeof(float);
  float*  featsF = (float*)p;  p += (size_t)RWS * DM * sizeof(float);    // 16 MiB
  __bf16* featsB = (__bf16*)p; p += (size_t)RWS * DM * sizeof(__bf16);   //  8 MiB
  __bf16* hidB   = (__bf16*)p; p += (size_t)RWS * FF * sizeof(__bf16);   // 32 MiB
  float*  ffnF   = (float*)p;                                            // 16 MiB

  // ---- 0) prep: fragment-swizzle activations & weights (bf16) ----
  cvt_frag_kernel<<<dim3((RWS * DM) / 256), 256, 0, stream>>>(x, xf, DM, (long long)RWS * DM);
  cvt_frag_kernel<<<dim3((1024 * DM) / 256), 256, 0, stream>>>(ipw, wcatf, DM, 1024LL * DM);
  cvt_frag_kernel<<<dim3((DM * DM) / 256), 256, 0, stream>>>(wq, wcatf + (size_t)64 * 16 * 512, DM, (long long)DM * DM);
  cvt_frag_kernel<<<dim3((DM * DM) / 256), 256, 0, stream>>>(wk, wcatf + (size_t)96 * 16 * 512, DM, (long long)DM * DM);
  cvt_frag_kernel<<<dim3((DM * DM) / 256), 256, 0, stream>>>(wo, wcatf + (size_t)128 * 16 * 512, DM, (long long)DM * DM);
  cvt_frag_kernel<<<dim3((FF * DM) / 256), 256, 0, stream>>>(w1, w1f, DM, (long long)FF * DM);
  cvt_frag_kernel<<<dim3((DM * FF) / 256), 256, 0, stream>>>(w2, w2f, FF, (long long)DM * FF);
  bias_cat_kernel<<<dim3((5 * DM + 255) / 256), 256, 0, stream>>>(ipb, bq, bk, bo, bcat);

  // ---- 1) projections: 512 x 160 tiles, 1 wave each ----
  proj_kernel<<<dim3((RWS / 16) * ((5 * DM) / 16) / 4), 128, 0, stream>>>(xf, wcatf, bcat, projsw);

  // ---- 2) fused attention: 4096 q-tiles, 1 wave per block ----
  attn_kernel<<<dim3(4096), 32, 0, stream>>>(am, prior, Qs, Ks, PQs, PKs, Vs,
                                             out_sa, out_pn, out_as, featsF);

  // ---- 3) LN1 (+ fragment-layout bf16 copy for FFN1) ----
  ln_kernel<<<dim3(RWS / 8), 256, 0, stream>>>(featsF, x, g1, be1, featsF, featsB);

  // ---- 4) FFN1 + exact GELU ----
  ffn1_kernel<<<dim3((RWS / 16) * (FF / 16) / 4), 128, 0, stream>>>(featsB, w1f, b1, hidB);

  // ---- 5) FFN2 ----
  ffn2_kernel<<<dim3((RWS / 16) * (DM / 16) / 4), 128, 0, stream>>>(hidB, w2f, b2, ffnF);

  // ---- 6) LN2 -> d_out ----
  ln_kernel<<<dim3(RWS / 8), 256, 0, stream>>>(featsF, ffnF, g2, be2, out, (__bf16*)nullptr);
}
